// LSTM_53618371723786
// MI455X (gfx1250) — matrix operand order repaired and checked by reading, not compiled, not run on hardware
//
#include <hip/hip_runtime.h>
#include <hip/hip_bf16.h>

typedef __attribute__((ext_vector_type(16))) _Float16 v16h;
typedef __attribute__((ext_vector_type(8)))  float    v8f;

union FragU { uint4 u[2]; v16h h; };
union HalfPack { _Float16 h[2]; unsigned u; };

#define Bsz 8192
#define Tsz 24
#define Hsz 1024
#define LDSH 72                   // padded LDS row stride in halves (144B)
#define BUFH (256 * LDSH)         // halves per LDS buffer

__device__ __forceinline__ float sigm_(float x) { return 1.0f / (1.0f + __expf(-x)); }
__device__ __forceinline__ float tanh_(float x) { float e = __expf(2.0f * x); return 1.0f - 2.0f / (e + 1.0f); }

// ---------------------------------------------------------------------------
// Pack W[4H][H] (row-major f32) into WMMA B-fragment order (f16).
// Fragment f = ((g*64 + j)*32 + q): 32x16 tile, cols n=j*16..+16, K=q*32..+32.
// Dword idx in frag = h*128 + lane*4 + d ; holds K = q*32 + h*16 + (lane<16?0:8) + 2*d, +1
// ---------------------------------------------------------------------------
__global__ void pack_w_kernel(const float* __restrict__ W, unsigned* __restrict__ dst) {
    unsigned idx  = blockIdx.x * 256u + threadIdx.x;          // 0 .. 2M-1
    unsigned frag = idx >> 8;
    unsigned inner = idx & 255u;
    unsigned hh  = inner >> 7;
    unsigned rem = inner & 127u;
    unsigned l = rem >> 2, d = rem & 3u;
    unsigned q = frag & 31u;
    unsigned j = (frag >> 5) & 63u;
    unsigned g = frag >> 11;
    unsigned n = j * 16u + (l & 15u);
    unsigned k = q * 32u + hh * 16u + ((l < 16u) ? 0u : 8u) + 2u * d;
    const float* row = W + (size_t)(g * Hsz + n) * Hsz;
    HalfPack cv;
    cv.h[0] = (_Float16)row[k];
    cv.h[1] = (_Float16)row[k + 1];
    dst[idx] = cv.u;
}

// ---------------------------------------------------------------------------
// One-time state/bias prep.
// ---------------------------------------------------------------------------
__global__ void prep_kernel(const float* __restrict__ h1_0, const float* __restrict__ c1_0,
                            const float* __restrict__ h2_0, const float* __restrict__ c2_0,
                            const float* __restrict__ b_ih1, const float* __restrict__ b_hh1,
                            const float* __restrict__ b_ih2, const float* __restrict__ b_hh2,
                            const float* __restrict__ W_ih1,
                            _Float16* __restrict__ h1a, _Float16* __restrict__ h2a,
                            float* __restrict__ c1, float* __restrict__ c2,
                            float* __restrict__ h2f,
                            float* __restrict__ bias1, float* __restrict__ bias2,
                            float* __restrict__ wih1v) {
    size_t i = (size_t)blockIdx.x * 256 + threadIdx.x;
    if (i < (size_t)Bsz * Hsz) {
        h1a[i] = (_Float16)h1_0[i];
        h2a[i] = (_Float16)h2_0[i];
        c1[i]  = c1_0[i];
        c2[i]  = c2_0[i];
        h2f[i] = h2_0[i];
    }
    if (i < 4 * Hsz) {
        bias1[i] = b_ih1[i] + b_hh1[i];
        bias2[i] = b_ih2[i] + b_hh2[i];
        wih1v[i] = W_ih1[i];   // W_ih1 is [4H,1]
    }
}

// ---------------------------------------------------------------------------
// GEMM pass: acc[tile][gate] += hsrc[bm..bm+256][:] @ Wpack(gate, n-tile jn).
// Double-buffered LDS A tiles (256x64 halves, stride 72): next chunk's global
// loads are issued before computing the current chunk; 1 barrier per chunk.
// Each wave owns 2 M-subtiles (rows wv*32 + {0,16}); each B fragment feeds
// 2 WMMAs (halves B-operand L2 traffic per WMMA).
// ---------------------------------------------------------------------------
__device__ __forceinline__ void gemm_pass(const _Float16* __restrict__ hsrc,
                                          const uint4* __restrict__ wpack,
                                          int bm, int jn, int wv, int lane, int tid,
                                          _Float16* lds, v8f acc[2][4]) {
    uint4 r[8];
    // prologue: stage chunk 0
    #pragma unroll
    for (int i = 0; i < 8; ++i) {
        int seg = tid + i * 256, row = seg >> 3, cs = seg & 7;
        r[i] = *(const uint4*)(hsrc + (size_t)(bm + row) * Hsz + cs * 8);
    }
    #pragma unroll
    for (int i = 0; i < 8; ++i) {
        int seg = tid + i * 256, row = seg >> 3, cs = seg & 7;
        *(uint4*)(lds + row * LDSH + cs * 8) = r[i];
    }
    __syncthreads();

    int cur = 0;
    for (int kc = 0; kc < Hsz; kc += 64) {
        bool more = (kc + 64) < Hsz;
        if (more) {
            #pragma unroll
            for (int i = 0; i < 8; ++i) {
                int seg = tid + i * 256, row = seg >> 3, cs = seg & 7;
                r[i] = *(const uint4*)(hsrc + (size_t)(bm + row) * Hsz + (kc + 64) + cs * 8);
            }
        }
        _Float16* lbase = lds + cur * BUFH;
        #pragma unroll
        for (int s = 0; s < 2; ++s) {
            int m0    = wv * 32 + (lane & 15);
            int khalf = (lane >> 4) * 8;
            const uint4* ap0 = (const uint4*)(lbase + m0 * LDSH + s * 32 + khalf);
            const uint4* ap1 = (const uint4*)(lbase + (m0 + 16) * LDSH + s * 32 + khalf);
            FragU a0, a1;
            a0.u[0] = ap0[0]; a0.u[1] = ap0[2];   // K khalf+0..7, khalf+16..23
            a1.u[0] = ap1[0]; a1.u[1] = ap1[2];
            int q = (kc >> 5) + s;
            #pragma unroll
            for (int g = 0; g < 4; ++g) {
                const uint4* bp = wpack + (size_t)((g * 64 + jn) * 32 + q) * 64 + lane;
                FragU bf;
                bf.u[0] = bp[0];
                bf.u[1] = bp[32];
                acc[0][g] = __builtin_amdgcn_wmma_f32_16x16x32_f16(
                    false, a0.h, false, bf.h, (short)0, acc[0][g], false, false);
                acc[1][g] = __builtin_amdgcn_wmma_f32_16x16x32_f16(
                    false, a1.h, false, bf.h, (short)0, acc[1][g], false, false);
            }
        }
        if (more) {
            _Float16* nbase = lds + (cur ^ 1) * BUFH;
            #pragma unroll
            for (int i = 0; i < 8; ++i) {
                int seg = tid + i * 256, row = seg >> 3, cs = seg & 7;
                *(uint4*)(nbase + row * LDSH + cs * 8) = r[i];
            }
        }
        __syncthreads();
        cur ^= 1;
    }
}

// ---------------------------------------------------------------------------
// Layer 1: gates = h1_prev @ W_hh1^T + x_t * W_ih1 + (b_ih1+b_hh1); LSTM cell.
// grid = (B/256, H/16), block = 256 (8 waves).
// ---------------------------------------------------------------------------
__global__ __launch_bounds__(256) void lstm_l1_kernel(
        const _Float16* __restrict__ h1_in, _Float16* __restrict__ h1_out,
        float* __restrict__ c1, const float* __restrict__ x, int t,
        const uint4* __restrict__ whh1, const float* __restrict__ bias1,
        const float* __restrict__ wih1v) {
    __shared__ __align__(16) _Float16 lds[2 * BUFH];
    int tid = threadIdx.x, wv = tid >> 5, lane = tid & 31;
    int bm = blockIdx.x * 256, jn = blockIdx.y;

    v8f acc[2][4];
    #pragma unroll
    for (int tl = 0; tl < 2; ++tl)
        #pragma unroll
        for (int g = 0; g < 4; ++g)
            #pragma unroll
            for (int r = 0; r < 8; ++r) acc[tl][g][r] = 0.0f;

    gemm_pass(h1_in, whh1, bm, jn, wv, lane, tid, lds, acc);

    int n = jn * 16 + (lane & 15);
    float bi = bias1[n],            bf = bias1[Hsz + n];
    float bg = bias1[2 * Hsz + n],  bo = bias1[3 * Hsz + n];
    float wi = wih1v[n],            wf = wih1v[Hsz + n];
    float wg = wih1v[2 * Hsz + n],  wo = wih1v[3 * Hsz + n];
    #pragma unroll
    for (int tl = 0; tl < 2; ++tl) {
        #pragma unroll
        for (int r = 0; r < 8; ++r) {
            int mrow = bm + wv * 32 + tl * 16 + r + ((lane >> 4) << 3);
            float xv = x[(size_t)mrow * Tsz + t];
            float gi = acc[tl][0][r] + xv * wi + bi;
            float gf = acc[tl][1][r] + xv * wf + bf;
            float gg = acc[tl][2][r] + xv * wg + bg;
            float go = acc[tl][3][r] + xv * wo + bo;
            size_t p = (size_t)mrow * Hsz + n;
            float cn = sigm_(gf) * c1[p] + sigm_(gi) * tanh_(gg);
            float hn = sigm_(go) * tanh_(cn);
            c1[p] = cn;
            h1_out[p] = (_Float16)hn;
        }
    }
}

// ---------------------------------------------------------------------------
// Layer 2: gates = h1_cur @ W_ih2^T + h2_prev @ W_hh2^T + bias2; LSTM cell.
// ---------------------------------------------------------------------------
__global__ __launch_bounds__(256) void lstm_l2_kernel(
        const _Float16* __restrict__ h1_cur, const _Float16* __restrict__ h2_in,
        _Float16* __restrict__ h2_out, float* __restrict__ h2f32,
        float* __restrict__ c2,
        const uint4* __restrict__ wih2, const uint4* __restrict__ whh2,
        const float* __restrict__ bias2) {
    __shared__ __align__(16) _Float16 lds[2 * BUFH];
    int tid = threadIdx.x, wv = tid >> 5, lane = tid & 31;
    int bm = blockIdx.x * 256, jn = blockIdx.y;

    v8f acc[2][4];
    #pragma unroll
    for (int tl = 0; tl < 2; ++tl)
        #pragma unroll
        for (int g = 0; g < 4; ++g)
            #pragma unroll
            for (int r = 0; r < 8; ++r) acc[tl][g][r] = 0.0f;

    gemm_pass(h1_cur, wih2, bm, jn, wv, lane, tid, lds, acc);
    gemm_pass(h2_in,  whh2, bm, jn, wv, lane, tid, lds, acc);

    int n = jn * 16 + (lane & 15);
    float bi = bias2[n],           bf = bias2[Hsz + n];
    float bg = bias2[2 * Hsz + n], bo = bias2[3 * Hsz + n];
    #pragma unroll
    for (int tl = 0; tl < 2; ++tl) {
        #pragma unroll
        for (int r = 0; r < 8; ++r) {
            int mrow = bm + wv * 32 + tl * 16 + r + ((lane >> 4) << 3);
            float gi = acc[tl][0][r] + bi;
            float gf = acc[tl][1][r] + bf;
            float gg = acc[tl][2][r] + bg;
            float go = acc[tl][3][r] + bo;
            size_t p = (size_t)mrow * Hsz + n;
            float cn = sigm_(gf) * c2[p] + sigm_(gi) * tanh_(gg);
            float hn = sigm_(go) * tanh_(cn);
            c2[p] = cn;
            h2_out[p] = (_Float16)hn;
            h2f32[p]  = hn;
        }
    }
}

// ---------------------------------------------------------------------------
// Final projection: out[b,:2] = h2 @ W_out^T + b_out. One wave per row.
// ---------------------------------------------------------------------------
__global__ __launch_bounds__(256) void out_proj_kernel(
        const float* __restrict__ h2, const float* __restrict__ W_out,
        const float* __restrict__ b_out, float* __restrict__ out) {
    int wv = threadIdx.x >> 5, lane = threadIdx.x & 31;
    int row = blockIdx.x * 8 + wv;
    const float* hr = h2 + (size_t)row * Hsz;
    float s0 = 0.0f, s1 = 0.0f;
    for (int k = lane; k < Hsz; k += 32) {
        float v = hr[k];
        s0 += v * W_out[k];
        s1 += v * W_out[Hsz + k];
    }
    #pragma unroll
    for (int off = 16; off > 0; off >>= 1) {
        s0 += __shfl_down(s0, off, 32);
        s1 += __shfl_down(s1, off, 32);
    }
    if (lane == 0) {
        out[(size_t)row * 2 + 0] = s0 + b_out[0];
        out[(size_t)row * 2 + 1] = s1 + b_out[1];
    }
}

// ---------------------------------------------------------------------------
extern "C" void kernel_launch(void* const* d_in, const int* in_sizes, int n_in,
                              void* d_out, int out_size, void* d_ws, size_t ws_size,
                              hipStream_t stream) {
    const float* x     = (const float*)d_in[0];
    const float* h1_0  = (const float*)d_in[1];
    const float* c1_0  = (const float*)d_in[2];
    const float* h2_0  = (const float*)d_in[3];
    const float* c2_0  = (const float*)d_in[4];
    const float* W_ih1 = (const float*)d_in[5];
    const float* W_hh1 = (const float*)d_in[6];
    const float* b_ih1 = (const float*)d_in[7];
    const float* b_hh1 = (const float*)d_in[8];
    const float* W_ih2 = (const float*)d_in[9];
    const float* W_hh2 = (const float*)d_in[10];
    const float* b_ih2 = (const float*)d_in[11];
    const float* b_hh2 = (const float*)d_in[12];
    const float* W_out = (const float*)d_in[13];
    const float* b_out = (const float*)d_in[14];
    float* out = (float*)d_out;

    const size_t MB = 1ull << 20;
    char* ws = (char*)d_ws;
    unsigned* whh1_pack = (unsigned*)(ws + 0 * MB);       // 8 MB each
    unsigned* wih2_pack = (unsigned*)(ws + 8 * MB);
    unsigned* whh2_pack = (unsigned*)(ws + 16 * MB);
    float*    bias1     = (float*)(ws + 24 * MB);
    float*    bias2     = (float*)(ws + 24 * MB + 16384);
    float*    wih1v     = (float*)(ws + 24 * MB + 32768);
    _Float16* h1buf[2]  = { (_Float16*)(ws + 25 * MB), (_Float16*)(ws + 41 * MB) };
    _Float16* h2buf[2]  = { (_Float16*)(ws + 57 * MB), (_Float16*)(ws + 73 * MB) };
    float*    c1        = (float*)(ws + 89 * MB);
    float*    c2        = (float*)(ws + 121 * MB);
    float*    h2f32     = (float*)(ws + 153 * MB);        // end: 185 MB

    // ---- one-time weight packing + state init ----
    pack_w_kernel<<<8192, 256, 0, stream>>>(W_hh1, whh1_pack);
    pack_w_kernel<<<8192, 256, 0, stream>>>(W_ih2, wih2_pack);
    pack_w_kernel<<<8192, 256, 0, stream>>>(W_hh2, whh2_pack);
    prep_kernel<<<(Bsz * Hsz) / 256, 256, 0, stream>>>(
        h1_0, c1_0, h2_0, c2_0, b_ih1, b_hh1, b_ih2, b_hh2, W_ih1,
        h1buf[0], h2buf[0], c1, c2, h2f32, bias1, bias2, wih1v);

    // ---- recurrence ----
    dim3 grid(Bsz / 256, Hsz / 16);
    for (int t = 0; t < Tsz; ++t) {
        int in = t & 1, o = in ^ 1;
        lstm_l1_kernel<<<grid, 256, 0, stream>>>(
            h1buf[in], h1buf[o], c1, x, t, (const uint4*)whh1_pack, bias1, wih1v);
        lstm_l2_kernel<<<grid, 256, 0, stream>>>(
            h1buf[o], h2buf[in], h2buf[o], h2f32, c2,
            (const uint4*)wih2_pack, (const uint4*)whh2_pack, bias2);
    }

    // ---- output projection ----
    out_proj_kernel<<<Bsz / 8, 256, 0, stream>>>(h2f32, W_out, b_out, out);
}
